// RainMamba_82197084111296
// MI455X (gfx1250) — matrix operand order repaired and checked by reading, not compiled, not run on hardware
//
#include <hip/hip_runtime.h>
#include <hip/hip_bf16.h>

typedef __attribute__((ext_vector_type(16))) _Float16 v16h;
typedef __attribute__((ext_vector_type(8)))  _Float16 v8h;
typedef __attribute__((ext_vector_type(8)))  float    v8f;

static inline int cdiv(long a, long b) { return (int)((a + b - 1) / b); }

__device__ __forceinline__ float act_apply(float x, int act) {
  switch (act) {
    case 1: return x > 0.f ? x : 0.f;                                   // relu
    case 2: return 0.5f * x * (1.f + erff(x * 0.7071067811865475f));    // gelu (exact)
    case 3: return x / (1.f + __expf(-x));                              // silu
    case 4: { const float al = 1.6732632423543772f, sc = 1.0507009873554805f;
              return x > 0.f ? sc * x : sc * al * (__expf(x) - 1.f); }  // selu
    case 5: return x > 20.f ? x : log1pf(__expf(x));                    // softplus
    default: return x;
  }
}

// ---------------------------------------------------------------------------
// WMMA GEMM: C(MxN) = act(A(MxK) * B(KxN) + bias), fp32 in/out, f16 compute.
// 128x64 block tile, BK=32, 256 threads = 8 waves in a 4(m) x 2(n) grid;
// each wave computes a 32x32 output patch = 4 x v_wmma_f32_16x16x32_f16.
// Staging is branch-free: a uniform fast path with vectorized loads for
// interior tiles, and a clamp+select (v_cndmask) path for edge tiles.
// bias_mode: 0 none, 1 per-column(N), 2 per-row(M).
// ---------------------------------------------------------------------------
#define BM 128
#define BN 64
#define BK 32
#define LDP (BK + 16)   // LDS row pitch in halves: 48 halves = 96B (16B aligned)

struct __attribute__((packed, aligned(4))) f4u { float x, y, z, w; };

__global__ __launch_bounds__(256) void gemm_f32_wmma(
    const float* __restrict__ A, int lda,
    const float* __restrict__ B, int ldb,
    float* __restrict__ C, int ldc,
    int M, int N, int K,
    const float* __restrict__ bias, int bias_mode, int act)
{
  __shared__ _Float16 As[BM][LDP];
  __shared__ _Float16 Bs[BN][LDP];

  const int tid  = threadIdx.x;
  const int lane = tid & 31;
  const int wv   = tid >> 5;
  const int m0   = blockIdx.y * BM;
  const int n0   = blockIdx.x * BN;

  const int mg = (wv & 3) * 16;     // wave m-stripes: mg and mg+64
  const int ng = (wv >> 2) * 32;    // wave n-stripes: ng and ng+16

  v8f acc00 = {}, acc01 = {}, acc10 = {}, acc11 = {};

  // A staging: row = tid>>1 (0..127), 16 consecutive k at (tid&1)*16
  const int ar = tid >> 1;
  const int ac = (tid & 1) * 16;
  // B staging: n = tid&63 (lane-contiguous), k-octet = (tid>>6)*8
  const int bn  = tid & 63;
  const int bks = (tid >> 6) * 8;

  const bool mn_full = (m0 + BM <= M) && (n0 + BN <= N);

  for (int k0 = 0; k0 < K; k0 += BK) {
    if (mn_full && (k0 + BK <= K)) {
      // ---- fast path: unconditional vectorized staging ----
      const float* ap = &A[(size_t)(m0 + ar) * lda + (k0 + ac)];
      const f4u a0 = *(const f4u*)(ap + 0);
      const f4u a1 = *(const f4u*)(ap + 4);
      const f4u a2 = *(const f4u*)(ap + 8);
      const f4u a3 = *(const f4u*)(ap + 12);
      v8h h0 = { (_Float16)a0.x, (_Float16)a0.y, (_Float16)a0.z, (_Float16)a0.w,
                 (_Float16)a1.x, (_Float16)a1.y, (_Float16)a1.z, (_Float16)a1.w };
      v8h h1 = { (_Float16)a2.x, (_Float16)a2.y, (_Float16)a2.z, (_Float16)a2.w,
                 (_Float16)a3.x, (_Float16)a3.y, (_Float16)a3.z, (_Float16)a3.w };
      *(v8h*)&As[ar][ac]     = h0;
      *(v8h*)&As[ar][ac + 8] = h1;

      const float* bp = &B[(size_t)(k0 + bks) * ldb + (n0 + bn)];
      const float b0 = bp[0];
      const float b1 = bp[(size_t)1 * ldb];
      const float b2 = bp[(size_t)2 * ldb];
      const float b3 = bp[(size_t)3 * ldb];
      const float b4 = bp[(size_t)4 * ldb];
      const float b5 = bp[(size_t)5 * ldb];
      const float b6 = bp[(size_t)6 * ldb];
      const float b7 = bp[(size_t)7 * ldb];
      v8h hb = { (_Float16)b0, (_Float16)b1, (_Float16)b2, (_Float16)b3,
                 (_Float16)b4, (_Float16)b5, (_Float16)b6, (_Float16)b7 };
      *(v8h*)&Bs[bn][bks] = hb;

      if (k0 + BK < K) {
        __builtin_prefetch(ap + BK, 0, 1);
        __builtin_prefetch(bp + (size_t)BK * ldb, 0, 1);
      }
    } else {
      // ---- edge path: clamp index + select, no divergent branches ----
      const int gm  = m0 + ar;
      const int gmc = gm < M ? gm : M - 1;
      #pragma unroll
      for (int j = 0; j < 16; ++j) {
        const int gk  = k0 + ac + j;
        const int gkc = gk < K ? gk : K - 1;
        const bool ok = (gm < M) && (gk < K);
        const float v = A[(size_t)gmc * lda + gkc];
        As[ar][ac + j] = ok ? (_Float16)v : (_Float16)0.f;
      }
      const int gn  = n0 + bn;
      const int gnc = gn < N ? gn : N - 1;
      #pragma unroll
      for (int j = 0; j < 8; ++j) {
        const int gk  = k0 + bks + j;
        const int gkc = gk < K ? gk : K - 1;
        const bool ok = (gk < K) && (gn < N);
        const float v = B[(size_t)gkc * ldb + gnc];
        Bs[bn][bks + j] = ok ? (_Float16)v : (_Float16)0.f;
      }
    }
    __syncthreads();

    // Fragments. A (16x32 f16): lane<16 -> row M=lane, halves K0..7|K16..23;
    // lane>=16 -> row M=lane-16, halves K8..15|K24..31.
    // B (32x16 f16): col = lane&15; lane<16 -> K0..7|K16..23, lane>=16 -> K8..15|K24..31.
    union FV { v16h v; v8h p[2]; };
    FV a0f, a1f, b0f, b1f;
    const int r0    = mg + (lane & 15);
    const int koff  = (lane < 16) ? 0 : 8;
    a0f.p[0] = *(const v8h*)&As[r0][koff];
    a0f.p[1] = *(const v8h*)&As[r0][koff + 16];
    a1f.p[0] = *(const v8h*)&As[r0 + 64][koff];
    a1f.p[1] = *(const v8h*)&As[r0 + 64][koff + 16];
    const int bc0   = ng + (lane & 15);
    const int bkoff = (lane < 16) ? 0 : 8;
    b0f.p[0] = *(const v8h*)&Bs[bc0][bkoff];
    b0f.p[1] = *(const v8h*)&Bs[bc0][bkoff + 16];
    b1f.p[0] = *(const v8h*)&Bs[bc0 + 16][bkoff];
    b1f.p[1] = *(const v8h*)&Bs[bc0 + 16][bkoff + 16];

    acc00 = __builtin_amdgcn_wmma_f32_16x16x32_f16(false, a0f.v, false, b0f.v,
                                                   (short)0, acc00, false, false);
    acc01 = __builtin_amdgcn_wmma_f32_16x16x32_f16(false, a0f.v, false, b1f.v,
                                                   (short)0, acc01, false, false);
    acc10 = __builtin_amdgcn_wmma_f32_16x16x32_f16(false, a1f.v, false, b0f.v,
                                                   (short)0, acc10, false, false);
    acc11 = __builtin_amdgcn_wmma_f32_16x16x32_f16(false, a1f.v, false, b1f.v,
                                                   (short)0, acc11, false, false);
    __syncthreads();
  }

  // D layout: VGPR i -> M = stripe + i + (lane<16 ? 0 : 8), col = lane&15
  const int radd = (lane < 16) ? 0 : 8;
  const int c0 = n0 + ng + (lane & 15);
  const int c1 = c0 + 16;
  #pragma unroll
  for (int i = 0; i < 8; ++i) {
    const int ma = m0 + mg + radd + i;
    const int mb = ma + 64;
    if (ma < M) {
      if (c0 < N) {
        float v = acc00[i];
        if (bias_mode == 1) v += bias[c0]; else if (bias_mode == 2) v += bias[ma];
        C[(size_t)ma * ldc + c0] = act_apply(v, act);
      }
      if (c1 < N) {
        float v = acc01[i];
        if (bias_mode == 1) v += bias[c1]; else if (bias_mode == 2) v += bias[ma];
        C[(size_t)ma * ldc + c1] = act_apply(v, act);
      }
    }
    if (mb < M) {
      if (c0 < N) {
        float v = acc10[i];
        if (bias_mode == 1) v += bias[c0]; else if (bias_mode == 2) v += bias[mb];
        C[(size_t)mb * ldc + c0] = act_apply(v, act);
      }
      if (c1 < N) {
        float v = acc11[i];
        if (bias_mode == 1) v += bias[c1]; else if (bias_mode == 2) v += bias[mb];
        C[(size_t)mb * ldc + c1] = act_apply(v, act);
      }
    }
  }
}

// ---------------------------------------------------------------------------
// im2col (2D, single image, symmetric pad)
// ---------------------------------------------------------------------------
__global__ void im2col2d_k(const float* __restrict__ x, float* __restrict__ col,
                           int Cin, int Hc, int Wc, int KH, int KW,
                           int S, int P, int OH, int OW)
{
  const int L = OH * OW;
  const long total = (long)Cin * KH * KW * L;
  const long i = (long)blockIdx.x * blockDim.x + threadIdx.x;
  if (i >= total) return;
  const int l = (int)(i % L); long rest = i / L;
  const int kw = (int)(rest % KW); rest /= KW;
  const int kh = (int)(rest % KH); rest /= KH;
  const int ci = (int)rest;
  const int oy = l / OW, ox = l % OW;
  const int iy = oy * S - P + kh;
  const int ix = ox * S - P + kw;
  float v = 0.f;
  if (iy >= 0 && iy < Hc && ix >= 0 && ix < Wc)
    v = x[((size_t)ci * Hc + iy) * Wc + ix];
  col[i] = v;
}

// ---------------------------------------------------------------------------
// im2col (3D, arbitrary strides, low pads, lhs-dilation, output-col chunking)
// ---------------------------------------------------------------------------
__global__ void im2col3d_k(const float* __restrict__ x, float* __restrict__ col,
    int Cin, int D, int Hc, int Wc,
    long sC, long sD, long sH, long sW,
    int KD, int KH, int KW, int SD, int SH, int SW, int PD, int PH, int PW,
    int LDd, int LHd, int LWd, int OH, int OW, int l0, int Lc)
{
  const long total = (long)Cin * KD * KH * KW * Lc;
  const long i = (long)blockIdx.x * blockDim.x + threadIdx.x;
  if (i >= total) return;
  const int lc = (int)(i % Lc); long rest = i / Lc;
  const int kw = (int)(rest % KW); rest /= KW;
  const int kh = (int)(rest % KH); rest /= KH;
  const int kd = (int)(rest % KD); rest /= KD;
  const int ci = (int)rest;
  const int l = l0 + lc;
  const int ox = l % OW; const int tt = l / OW;
  const int oy = tt % OH; const int od = tt / OH;
  const int td = od * SD - PD + kd;
  const int ty = oy * SH - PH + kh;
  const int tx = ox * SW - PW + kw;
  float v = 0.f;
  if (td >= 0 && ty >= 0 && tx >= 0 &&
      (td % LDd) == 0 && (ty % LHd) == 0 && (tx % LWd) == 0) {
    const int idd = td / LDd, iyy = ty / LHd, ixx = tx / LWd;
    if (idd < D && iyy < Hc && ixx < Wc)
      v = x[(size_t)ci * sC + (size_t)idd * sD + (size_t)iyy * sH + (size_t)ixx * sW];
  }
  col[i] = v;
}

// ---------------------------------------------------------------------------
// LayerNorm over C with layout (outer, C, inner); one thread per (outer,inner)
// ---------------------------------------------------------------------------
__global__ void layernorm_k(const float* __restrict__ x, float* __restrict__ y,
                            const float* __restrict__ g, const float* __restrict__ b,
                            int outer, int C, int inner, float eps)
{
  const long total = (long)outer * inner;
  const long i = (long)blockIdx.x * blockDim.x + threadIdx.x;
  if (i >= total) return;
  const int o = (int)(i / inner), in = (int)(i % inner);
  const float* xp = x + (size_t)o * C * inner + in;
  float mu = 0.f;
  for (int c = 0; c < C; ++c) mu += xp[(size_t)c * inner];
  mu /= C;
  float var = 0.f;
  for (int c = 0; c < C; ++c) { float d = xp[(size_t)c * inner] - mu; var += d * d; }
  var /= C;
  const float inv = rsqrtf(var + eps);
  float* yp = y + (size_t)o * C * inner + in;
  for (int c = 0; c < C; ++c)
    yp[(size_t)c * inner] = (xp[(size_t)c * inner] - mu) * inv * g[c] + b[c];
}

// ---------------------------------------------------------------------------
// depthwise 7x7 conv pad 3 (single image)
// ---------------------------------------------------------------------------
__global__ void dwconv7_k(const float* __restrict__ x, const float* __restrict__ w,
                          const float* __restrict__ b, float* __restrict__ y,
                          int C, int Hc, int Wc)
{
  const long total = (long)C * Hc * Wc;
  const long i = (long)blockIdx.x * blockDim.x + threadIdx.x;
  if (i >= total) return;
  const int xw = (int)(i % Wc); long t = i / Wc;
  const int yy = (int)(t % Hc); const int c = (int)(t / Hc);
  const float* wp = w + (size_t)c * 49;
  float acc = b[c];
  for (int ky = 0; ky < 7; ++ky) {
    const int iy = yy + ky - 3; if (iy < 0 || iy >= Hc) continue;
    for (int kx = 0; kx < 7; ++kx) {
      const int ix = xw + kx - 3; if (ix < 0 || ix >= Wc) continue;
      acc += wp[ky * 7 + kx] * x[((size_t)c * Hc + iy) * Wc + ix];
    }
  }
  y[i] = acc;
}

__global__ void chw_to_hwc_k(const float* __restrict__ x, float* __restrict__ y,
                             int C, int HW)
{
  const long i = (long)blockIdx.x * blockDim.x + threadIdx.x;
  if (i >= (long)C * HW) return;
  const int c = (int)(i / HW), p = (int)(i % HW);
  y[(size_t)p * C + c] = x[i];
}

__global__ void hwc_scale_add_k(float* __restrict__ x, const float* __restrict__ y,
                                const float* __restrict__ gamma, int C, int HW)
{
  const long i = (long)blockIdx.x * blockDim.x + threadIdx.x;
  if (i >= (long)C * HW) return;
  const int c = (int)(i / HW), p = (int)(i % HW);
  x[i] += gamma[c] * y[(size_t)p * C + c];
}

__global__ void add_inplace_k(float* __restrict__ a, const float* __restrict__ b, long n)
{
  const long i = (long)blockIdx.x * blockDim.x + threadIdx.x;
  if (i < n) a[i] += b[i];
}

__global__ void avg4_k(const float* a, const float* b, const float* c,
                       const float* d, float* o, long n)
{
  const long i = (long)blockIdx.x * blockDim.x + threadIdx.x;
  if (i < n) o[i] = 0.25f * (a[i] + b[i] + c[i] + d[i]);
}

// align-corners bilinear resize (P planes)
__global__ void resize_ac_k(const float* __restrict__ x, float* __restrict__ y,
                            int P, int h, int w, int Ho, int Wo)
{
  const long total = (long)P * Ho * Wo;
  const long i = (long)blockIdx.x * blockDim.x + threadIdx.x;
  if (i >= total) return;
  const int xo = (int)(i % Wo); long t = i / Wo;
  const int yo = (int)(t % Ho); const int p = (int)(t / Ho);
  const float ysf = (Ho > 1) ? yo * (float)(h - 1) / (Ho - 1) : 0.f;
  const float xsf = (Wo > 1) ? xo * (float)(w - 1) / (Wo - 1) : 0.f;
  int y0 = (int)floorf(ysf); if (y0 > h - 1) y0 = h - 1; if (y0 < 0) y0 = 0;
  int x0 = (int)floorf(xsf); if (x0 > w - 1) x0 = w - 1; if (x0 < 0) x0 = 0;
  const int y1 = (y0 + 1 < h) ? y0 + 1 : h - 1;
  const int x1 = (x0 + 1 < w) ? x0 + 1 : w - 1;
  const float wy = ysf - y0, wx = xsf - x0;
  const float* xp = x + (size_t)p * h * w;
  y[i] = xp[(size_t)y0 * w + x0] * (1 - wy) * (1 - wx)
       + xp[(size_t)y0 * w + x1] * (1 - wy) * wx
       + xp[(size_t)y1 * w + x0] * wy * (1 - wx)
       + xp[(size_t)y1 * w + x1] * wy * wx;
}

// x5 (strided (t,c,h,w)) -> seq (L x C), L = (t*H + h)*W + w
__global__ void to_seq_k(const float* __restrict__ x5, float* __restrict__ seq,
                         int T_, int C_, int Hh, int Ww, long sT, long sC)
{
  const long total = (long)T_ * Hh * Ww * C_;
  const long i = (long)blockIdx.x * blockDim.x + threadIdx.x;
  if (i >= total) return;
  const int c = (int)(i % C_); const long l = i / C_;
  const int wq = (int)(l % Ww); long t2 = l / Ww;
  const int hq = (int)(t2 % Hh); const int tq = (int)(t2 / Hh);
  seq[(size_t)l * C_ + c] = x5[(size_t)tq * sT + (size_t)c * sC + (size_t)hq * Ww + wq];
}

__global__ void seq_add_x5_k(const float* __restrict__ y, float* __restrict__ x5,
                             int T_, int C_, int Hh, int Ww, long sT, long sC)
{
  const long total = (long)T_ * Hh * Ww * C_;
  const long i = (long)blockIdx.x * blockDim.x + threadIdx.x;
  if (i >= total) return;
  const int c = (int)(i % C_); const long l = i / C_;
  const int wq = (int)(l % Ww); long t2 = l / Ww;
  const int hq = (int)(t2 % Hh); const int tq = (int)(t2 / Hh);
  x5[(size_t)tq * sT + (size_t)c * sC + (size_t)hq * Ww + wq] += y[(size_t)l * C_ + c];
}

__global__ void gather_rows_k(const float* __restrict__ seq, float* __restrict__ out,
                              const int* __restrict__ idx, int L, int C_)
{
  const long total = (long)L * C_;
  const long i = (long)blockIdx.x * blockDim.x + threadIdx.x;
  if (i >= total) return;
  const int c = (int)(i % C_); const int r = (int)(i / C_);
  out[(size_t)r * C_ + c] = seq[(size_t)idx[r] * C_ + c];
}

// x5[curve[i]] += y[i]  (curve is a permutation -> race free)
__global__ void seq_scatter_add_x5_k(const float* __restrict__ y, float* __restrict__ x5,
                                     const int* __restrict__ idx,
                                     int T_, int C_, int Hh, int Ww, long sT, long sC)
{
  const long total = (long)T_ * Hh * Ww * C_;
  const long i = (long)blockIdx.x * blockDim.x + threadIdx.x;
  if (i >= total) return;
  const int c = (int)(i % C_); const int r = (int)(i / C_);
  const long l = idx[r];
  const int wq = (int)(l % Ww); long t2 = l / Ww;
  const int hq = (int)(t2 % Hh); const int tq = (int)(t2 / Hh);
  x5[(size_t)tq * sT + (size_t)c * sC + (size_t)hq * Ww + wq] += y[(size_t)r * C_ + c];
}

// causal depthwise conv1d (k=4) over L, from interleaved in_proj buffer + SiLU
__global__ void causal_dw_silu_k(const float* __restrict__ xin, int xstride,
                                 const float* __restrict__ w, const float* __restrict__ b,
                                 float* __restrict__ xs, int L, int di)
{
  const long total = (long)L * di;
  const long i = (long)blockIdx.x * blockDim.x + threadIdx.x;
  if (i >= total) return;
  const int d = (int)(i % di); const int l = (int)(i / di);
  float acc = b[d];
  #pragma unroll
  for (int j = 0; j < 4; ++j) {
    const int ll = l - 3 + j;
    if (ll >= 0) acc += w[(size_t)d * 4 + j] * xin[(size_t)ll * xstride + d];
  }
  xs[(size_t)l * di + d] = acc / (1.f + __expf(-acc));
}

// selective scan: one block, blockDim = d_inner; 16 states per thread in regs.
__global__ void mamba_scan_k(const float* __restrict__ dt, const float* __restrict__ xs,
                             const float* __restrict__ xp, int xpld, int boff,
                             const float* __restrict__ A_log, const float* __restrict__ Dp,
                             const float* __restrict__ z, int zstride,
                             float* __restrict__ y, int L, int di)
{
  const int d = threadIdx.x;
  __shared__ float Bs[16], Cs[16];
  float negA[16], h[16];
  #pragma unroll
  for (int n = 0; n < 16; ++n) { negA[n] = -__expf(A_log[(size_t)d * 16 + n]); h[n] = 0.f; }
  const float Dd = Dp[d];
  for (int l = 0; l < L; ++l) {
    __syncthreads();
    if (d < 16)       Bs[d]      = xp[(size_t)l * xpld + boff + d];
    else if (d < 32)  Cs[d - 16] = xp[(size_t)l * xpld + boff + 16 + (d - 16)];
    __syncthreads();
    const float dtv = dt[(size_t)l * di + d];
    const float xv  = xs[(size_t)l * di + d];
    const float dx  = dtv * xv;
    float acc = 0.f;
    #pragma unroll
    for (int n = 0; n < 16; ++n) {
      h[n] = __expf(dtv * negA[n]) * h[n] + dx * Bs[n];
      acc += h[n] * Cs[n];
    }
    const float zv = z[(size_t)l * zstride + d];
    y[(size_t)l * di + d] = (acc + xv * Dd) * (zv / (1.f + __expf(-zv)));
  }
}

// wf[co,ci,kd,kh,kw] = w[ci,co,KD-1-kd,KH-1-kh,KW-1-kw]
__global__ void tflip_w3d_k(const float* __restrict__ w, float* __restrict__ wf,
                            int Ci, int Co, int KD, int KH, int KW)
{
  const long total = (long)Ci * Co * KD * KH * KW;
  const long i = (long)blockIdx.x * blockDim.x + threadIdx.x;
  if (i >= total) return;
  const int kw = (int)(i % KW); long t = i / KW;
  const int kh = (int)(t % KH); t /= KH;
  const int kd = (int)(t % KD); t /= KD;
  const int ci = (int)(t % Ci); const int co = (int)(t / Ci);
  wf[i] = w[((((size_t)ci * Co + co) * KD + (KD - 1 - kd)) * KH + (KH - 1 - kh)) * KW
            + (KW - 1 - kw)];
}

// (Cc*4, T, h, w) -> (Cc, T, 2h, 2w)   (pixel shuffle r=2 in CTHW layout)
__global__ void pixel_shuffle2_k(const float* __restrict__ x, float* __restrict__ y,
                                 int Cc, int T_, int h, int w)
{
  const int Ho = 2 * h, Wo = 2 * w;
  const long total = (long)Cc * T_ * Ho * Wo;
  const long i = (long)blockIdx.x * blockDim.x + threadIdx.x;
  if (i >= total) return;
  const int xo = (int)(i % Wo); long t = i / Wo;
  const int yo = (int)(t % Ho); t /= Ho;
  const int tq = (int)(t % T_); const int c = (int)(t / T_);
  const int ci = c * 4 + (yo & 1) * 2 + (xo & 1);
  y[i] = x[(((size_t)ci * T_ + tq) * h + (yo >> 1)) * w + (xo >> 1)];
}

// (C,T,H,W) -> (T,C,H,W)
__global__ void cthw_to_tchw_k(const float* __restrict__ x, float* __restrict__ y,
                               int C, int T_, int HW)
{
  const long total = (long)C * T_ * HW;
  const long i = (long)blockIdx.x * blockDim.x + threadIdx.x;
  if (i >= total) return;
  const int p = (int)(i % HW); long t2 = i / HW;
  const int tq = (int)(t2 % T_); const int c = (int)(t2 / T_);
  y[((size_t)tq * C + c) * HW + p] = x[i];
}

// ---------------------------------------------------------------------------
// Host side
// ---------------------------------------------------------------------------
static void gemm(hipStream_t s, const float* A, int lda, const float* B, int ldb,
                 float* C, int ldc, int M, int N, int K,
                 const float* bias, int bmode, int act)
{
  dim3 grid(cdiv(N, BN), cdiv(M, BM));
  gemm_f32_wmma<<<grid, 256, 0, s>>>(A, lda, B, ldb, C, ldc, M, N, K, bias, bmode, act);
}

struct Ws {
  char* base; size_t cap; size_t off;
  float* alloc(size_t nfloats) {
    size_t bytes = (nfloats * sizeof(float) + 255) & ~(size_t)255;
    if (bytes > cap) bytes = cap;
    if (off + bytes > cap) off = 0;          // deterministic wrap
    float* p = (float*)(base + off); off += bytes; return p;
  }
};

struct MambaP {
  const float *norm_g, *norm_b, *in_w, *conv_w, *conv_b, *xproj_w,
              *dt_w, *dt_b, *A_log, *D, *out_w;
};
struct CNB { const float *dw_w, *dw_b, *ln_g, *ln_b, *w1, *b1, *w2, *b2, *gamma; };

extern "C" void kernel_launch(void* const* d_in, const int* in_sizes, int n_in,
                              void* d_out, int out_size, void* d_ws, size_t ws_size,
                              hipStream_t stream)
{
  (void)in_sizes; (void)n_in; (void)out_size;
  const float* lqs     = (const float*)d_in[0];
  const int*   curve_l = (const int*)d_in[1];
  const int*   curve_s = (const int*)d_in[2];

  int pci = 3;
  auto next = [&]() { return (const float*)d_in[pci++]; };

  Ws ws{(char*)d_ws, ws_size, 0};
  const int NT = 4;  // n*t images
  const int T_ = 4;
  const int dims[4]   = {96, 192, 384, 768};
  const int depths[4] = {3, 3, 9, 3};
  const int hsz[4]    = {32, 16, 8, 4};

  // ---- param cursor: convnext ----
  const float* stem_w = next(); const float* stem_b = next();
  const float* stem_ln_g = next(); const float* stem_ln_b = next();
  const float *ds_ln_g[4] = {}, *ds_ln_b[4] = {}, *ds_w[4] = {}, *ds_b[4] = {};
  CNB blocks[4][9];
  const float *out_ln_g[4], *out_ln_b[4];
  for (int i = 0; i < 4; ++i) {
    if (i > 0) { ds_ln_g[i] = next(); ds_ln_b[i] = next(); ds_w[i] = next(); ds_b[i] = next(); }
    for (int j = 0; j < depths[i]; ++j) {
      CNB& b = blocks[i][j];
      b.dw_w = next(); b.dw_b = next(); b.ln_g = next(); b.ln_b = next();
      b.w1 = next(); b.b1 = next(); b.w2 = next(); b.b2 = next(); b.gamma = next();
    }
    out_ln_g[i] = next(); out_ln_b[i] = next();
  }
  const float *head_w[4], *head_b[4];
  for (int i = 0; i < 4; ++i) { head_w[i] = next(); head_b[i] = next(); }
  const float* ref1_w = next(); const float* ref1_b = next();
  const float* ref2_w = next(); const float* ref2_b = next();
  const float* ref3_w = next(); const float* ref3_b = next();
  const float* conv1_w = next(); const float* conv1_b = next();
  const float* up2_w = next();  const float* up2_b = next();
  const float* cbu1_w = next(); const float* cbu1_b = next();
  const float* cbu2_w = next(); const float* cbu2_b = next();
  const float* last_w = next(); const float* last_b = next();
  auto read_mamba = [&](MambaP& m) {
    m.norm_g = next(); m.norm_b = next(); m.in_w = next(); m.conv_w = next();
    m.conv_b = next(); m.xproj_w = next(); m.dt_w = next(); m.dt_b = next();
    m.A_log = next(); m.D = next(); m.out_w = next();
  };
  MambaP mp128[8];  // g1,l1,g2,l2,g3,l3,g4,l4
  for (int i = 0; i < 8; ++i) read_mamba(mp128[i]);
  MambaP mp256[6];  // lg1,ll1,lg2,ll2,lg3,ll3
  for (int i = 0; i < 6; ++i) read_mamba(mp256[i]);

  // ---- helpers ----
  auto conv2d_full = [&](const float* xin, const float* wgt, const float* bias,
                         int Cin, int Hc, int KH, int S, int P, int OH,
                         int Cout, float* out, int act) {
    const int K = Cin * KH * KH, L = OH * OH;
    const size_t ck = ws.off;
    float* col = ws.alloc((size_t)K * L);
    im2col2d_k<<<cdiv((long)K * L, 256), 256, 0, stream>>>(xin, col, Cin, Hc, Hc,
                                                           KH, KH, S, P, OH, OH);
    gemm(stream, wgt, K, col, L, out, L, Cout, L, K, bias, 2, act);
    ws.off = ck;
  };

  auto conv3d_im2col = [&](const float* x, long sC, long sD, long sH, long sW,
                           int Cin, int D, int Hc, int Wc,
                           const float* wgt, const float* bias, int Cout,
                           int KD, int KH, int KW, int SD, int SH, int SW,
                           int PD, int PH, int PW, int LDd, int LHd, int LWd,
                           int OD, int OH, int OW, float* out, int act) {
    const int K = Cin * KD * KH * KW;
    const int Ltot = OD * OH * OW;
    int chunk = (int)(4194304 / K); if (chunk < 64) chunk = 64; if (chunk > Ltot) chunk = Ltot;
    const size_t ck = ws.off;
    float* col = ws.alloc((size_t)K * chunk);
    for (int l0 = 0; l0 < Ltot; l0 += chunk) {
      const int Lc = (chunk < Ltot - l0) ? chunk : (Ltot - l0);
      im2col3d_k<<<cdiv((long)K * Lc, 256), 256, 0, stream>>>(
          x, col, Cin, D, Hc, Wc, sC, sD, sH, sW, KD, KH, KW, SD, SH, SW,
          PD, PH, PW, LDd, LHd, LWd, OH, OW, l0, Lc);
      gemm(stream, wgt, K, col, Lc, out + l0, Ltot, Cout, Lc, K, bias, 2, act);
    }
    ws.off = ck;
  };

  auto cn_block_f = [&](float* x_img, const CNB& p, int C, int Hc) {
    const int HW = Hc * Hc;
    const size_t ck = ws.off;
    float* dw = ws.alloc((size_t)C * HW);
    dwconv7_k<<<cdiv((long)C * HW, 256), 256, 0, stream>>>(x_img, p.dw_w, p.dw_b, dw, C, Hc, Hc);
    float* hwc = ws.alloc((size_t)HW * C);
    chw_to_hwc_k<<<cdiv((long)C * HW, 256), 256, 0, stream>>>(dw, hwc, C, HW);
    layernorm_k<<<cdiv(HW, 128), 128, 0, stream>>>(hwc, hwc, p.ln_g, p.ln_b, HW, C, 1, 1e-6f);
    float* h1 = ws.alloc((size_t)HW * 4 * C);
    gemm(stream, hwc, C, p.w1, 4 * C, h1, 4 * C, HW, 4 * C, C, p.b1, 1, 2);  // gelu
    float* h2 = ws.alloc((size_t)HW * C);
    gemm(stream, h1, 4 * C, p.w2, C, h2, C, HW, C, 4 * C, p.b2, 1, 0);
    hwc_scale_add_k<<<cdiv((long)C * HW, 256), 256, 0, stream>>>(x_img, h2, p.gamma, C, HW);
    ws.off = ck;
  };

  auto mamba_run = [&](const MambaP& p, const float* seq_ln, int L, int dm, float* out) {
    const int di = 2 * dm, dtr = (dm + 15) / 16, xpn = dtr + 32;
    const size_t ck = ws.off;
    float* inb = ws.alloc((size_t)L * 2 * di);
    gemm(stream, seq_ln, dm, p.in_w, 2 * di, inb, 2 * di, L, 2 * di, dm, nullptr, 0, 0);
    float* xs = ws.alloc((size_t)L * di);
    causal_dw_silu_k<<<cdiv((long)L * di, 256), 256, 0, stream>>>(inb, 2 * di, p.conv_w,
                                                                  p.conv_b, xs, L, di);
    float* xp = ws.alloc((size_t)L * xpn);
    gemm(stream, xs, di, p.xproj_w, xpn, xp, xpn, L, xpn, di, nullptr, 0, 0);
    float* dt = ws.alloc((size_t)L * di);
    gemm(stream, xp, xpn, p.dt_w, di, dt, di, L, di, dtr, p.dt_b, 1, 5);     // softplus
    float* yb = ws.alloc((size_t)L * di);
    mamba_scan_k<<<1, di, 0, stream>>>(dt, xs, xp, xpn, dtr, p.A_log, p.D,
                                       inb + di, 2 * di, yb, L, di);
    gemm(stream, yb, di, p.out_w, dm, out, dm, L, dm, di, nullptr, 0, 0);
    ws.off = ck;
  };

  auto mamba_global_run = [&](const MambaP& p, float* x5b, int C_, int Hh, int Ww,
                              long sT, long sC) {
    const int L = T_ * Hh * Ww;
    const size_t ck = ws.off;
    float* seq = ws.alloc((size_t)L * C_);
    to_seq_k<<<cdiv((long)L * C_, 256), 256, 0, stream>>>(x5b, seq, T_, C_, Hh, Ww, sT, sC);
    layernorm_k<<<cdiv(L, 128), 128, 0, stream>>>(seq, seq, p.norm_g, p.norm_b, L, C_, 1, 1e-5f);
    float* out = ws.alloc((size_t)L * C_);
    mamba_run(p, seq, L, C_, out);
    seq_add_x5_k<<<cdiv((long)L * C_, 256), 256, 0, stream>>>(out, x5b, T_, C_, Hh, Ww, sT, sC);
    ws.off = ck;
  };

  auto mamba_local_run = [&](const MambaP& p, float* x5b, const int* curve,
                             int C_, int Hh, int Ww, long sT, long sC) {
    const int L = T_ * Hh * Ww;
    const size_t ck = ws.off;
    float* seq = ws.alloc((size_t)L * C_);
    to_seq_k<<<cdiv((long)L * C_, 256), 256, 0, stream>>>(x5b, seq, T_, C_, Hh, Ww, sT, sC);
    float* ps = ws.alloc((size_t)L * C_);
    gather_rows_k<<<cdiv((long)L * C_, 256), 256, 0, stream>>>(seq, ps, curve, L, C_);
    layernorm_k<<<cdiv(L, 128), 128, 0, stream>>>(ps, ps, p.norm_g, p.norm_b, L, C_, 1, 1e-5f);
    float* out = ws.alloc((size_t)L * C_);
    mamba_run(p, ps, L, C_, out);
    seq_scatter_add_x5_k<<<cdiv((long)L * C_, 256), 256, 0, stream>>>(out, x5b, curve,
                                                                      T_, C_, Hh, Ww, sT, sC);
    ws.off = ck;
  };

  // ================= ConvNeXt backbone =================
  float* x = ws.alloc((size_t)NT * 96 * 1024);
  for (int im = 0; im < NT; ++im)
    conv2d_full(lqs + (size_t)im * 3 * 128 * 128, stem_w, stem_b, 3, 128, 4, 4, 0, 32,
                96, x + (size_t)im * 96 * 1024, 0);
  layernorm_k<<<cdiv((long)NT * 1024, 256), 256, 0, stream>>>(x, x, stem_ln_g, stem_ln_b,
                                                              NT, 96, 1024, 1e-6f);
  float* feats[4];
  for (int st = 0; st < 4; ++st) {
    const int dimc = dims[st], Hn = hsz[st], HWn = Hn * Hn;
    if (st > 0) {
      const int dimp = dims[st - 1], Hp = hsz[st - 1], HWp = Hp * Hp;
      float* xn = ws.alloc((size_t)NT * dimc * HWn);
      const size_t ck = ws.off;
      float* lnb = ws.alloc((size_t)NT * dimp * HWp);
      layernorm_k<<<cdiv((long)NT * HWp, 256), 256, 0, stream>>>(x, lnb, ds_ln_g[st],
                                                                 ds_ln_b[st], NT, dimp, HWp, 1e-6f);
      for (int im = 0; im < NT; ++im)
        conv2d_full(lnb + (size_t)im * dimp * HWp, ds_w[st], ds_b[st], dimp, Hp, 2, 2, 0,
                    Hn, dimc, xn + (size_t)im * dimc * HWn, 0);
      ws.off = ck;
      x = xn;
    }
    for (int j = 0; j < depths[st]; ++j)
      for (int im = 0; im < NT; ++im)
        cn_block_f(x + (size_t)im * dimc * HWn, blocks[st][j], dimc, Hn);
    feats[st] = ws.alloc((size_t)NT * dimc * HWn);
    layernorm_k<<<cdiv((long)NT * HWn, 256), 256, 0, stream>>>(x, feats[st], out_ln_g[st],
                                                               out_ln_b[st], NT, dimc, HWn, 1e-6f);
  }

  // ================= heads + fusion =================
  float* downs[4];
  for (int i = 0; i < 4; ++i) {
    const int hw2 = hsz[i] * hsz[i];
    downs[i] = ws.alloc((size_t)NT * 128 * hw2);
    for (int im = 0; im < NT; ++im)
      gemm(stream, head_w[i], dims[i], feats[i] + (size_t)im * dims[i] * hw2, hw2,
           downs[i] + (size_t)im * 128 * hw2, hw2, 128, hw2, dims[i], head_b[i], 2, 0);
  }
  float* f = ws.alloc((size_t)NT * 128 * 1024);
  {
    const size_t ck = ws.off;
    float* r1 = ws.alloc((size_t)NT * 128 * 1024);
    float* r2 = ws.alloc((size_t)NT * 128 * 1024);
    float* r3 = ws.alloc((size_t)NT * 128 * 1024);
    resize_ac_k<<<cdiv((long)NT * 128 * 1024, 256), 256, 0, stream>>>(downs[1], r1, NT * 128, 16, 16, 32, 32);
    resize_ac_k<<<cdiv((long)NT * 128 * 1024, 256), 256, 0, stream>>>(downs[2], r2, NT * 128, 8, 8, 32, 32);
    resize_ac_k<<<cdiv((long)NT * 128 * 1024, 256), 256, 0, stream>>>(downs[3], r3, NT * 128, 4, 4, 32, 32);
    avg4_k<<<cdiv((long)NT * 128 * 1024, 256), 256, 0, stream>>>(downs[0], r1, r2, r3, f,
                                                                 (long)NT * 128 * 1024);
    ws.off = ck;
  }

  // refine head + residual -> x5 (T,C=128,32,32)
  float* x5 = ws.alloc((size_t)T_ * 128 * 1024);
  {
    const size_t ck = ws.off;
    float* r  = ws.alloc((size_t)NT * 128 * 1024);
    float* rr = ws.alloc((size_t)NT * 128 * 1024);
    for (int im = 0; im < NT; ++im)
      conv2d_full(f + (size_t)im * 131072, ref1_w, ref1_b, 128, 32, 3, 1, 1, 32, 128,
                  r + (size_t)im * 131072, 4);
    for (int im = 0; im < NT; ++im)
      conv2d_full(r + (size_t)im * 131072, ref2_w, ref2_b, 128, 32, 3, 1, 1, 32, 128,
                  rr + (size_t)im * 131072, 4);
    for (int im = 0; im < NT; ++im)
      gemm(stream, ref3_w, 128, rr + (size_t)im * 131072, 1024,
           x5 + (size_t)im * 131072, 1024, 128, 1024, 128, ref3_b, 2, 0);
    add_inplace_k<<<cdiv((long)NT * 131072, 256), 256, 0, stream>>>(x5, f, (long)NT * 131072);
    ws.off = ck;
  }

  // ================= M1: mamba @ 128, 32x32 (TCHW) =================
  mamba_global_run(mp128[0], x5, 128, 32, 32, 131072, 1024);
  mamba_local_run (mp128[1], x5, curve_l, 128, 32, 32, 131072, 1024);
  mamba_global_run(mp128[2], x5, 128, 32, 32, 131072, 1024);
  mamba_local_run (mp128[3], x5, curve_l, 128, 32, 32, 131072, 1024);

  // conv1: (1,128,4,32,32) -> (1,256,4,16,16), relu  (output CTHW)
  float* xd = ws.alloc((size_t)256 * 4 * 16 * 16);
  conv3d_im2col(x5, 1024, 131072, 32, 1, 128, 4, 32, 32, conv1_w, conv1_b, 256,
                3, 3, 3, 1, 2, 2, 1, 1, 1, 1, 1, 1, 4, 16, 16, xd, 1);

  // ================= M2: mamba @ 256, 16x16 (CTHW) =================
  mamba_global_run(mp256[0], xd, 256, 16, 16, 256, 1024);
  mamba_local_run (mp256[1], xd, curve_s, 256, 16, 16, 256, 1024);
  mamba_global_run(mp256[2], xd, 256, 16, 16, 256, 1024);
  mamba_local_run (mp256[3], xd, curve_s, 256, 16, 16, 256, 1024);
  mamba_global_run(mp256[4], xd, 256, 16, 16, 256, 1024);
  mamba_local_run (mp256[5], xd, curve_s, 256, 16, 16, 256, 1024);

  // up2: conv_transpose3d -> (1,128,4,32,32), relu (CTHW)
  float* xu = ws.alloc((size_t)128 * 4 * 32 * 32);
  {
    float* wf = ws.alloc((size_t)128 * 256 * 27);
    tflip_w3d_k<<<cdiv((long)128 * 256 * 27, 256), 256, 0, stream>>>(up2_w, wf, 256, 128, 3, 3, 3);
    conv3d_im2col(xd, 1024, 256, 16, 1, 256, 4, 16, 16, wf, up2_b, 128,
                  3, 3, 3, 1, 1, 1, 1, 1, 1, 1, 2, 2, 4, 32, 32, xu, 1);
  }

  // ================= M3: mamba @ 128, 32x32 (CTHW) =================
  mamba_global_run(mp128[4], xu, 128, 32, 32, 1024, 4096);
  mamba_local_run (mp128[5], xu, curve_l, 128, 32, 32, 1024, 4096);
  mamba_global_run(mp128[6], xu, 128, 32, 32, 1024, 4096);
  mamba_local_run (mp128[7], xu, curve_l, 128, 32, 32, 1024, 4096);

  // ================= upsampling tail =================
  float* xr1 = ws.alloc((size_t)128 * 4 * 32 * 32);
  conv3d_im2col(xu, 4096, 1024, 32, 1, 128, 4, 32, 32, cbu1_w, cbu1_b, 128,
                1, 3, 3, 1, 1, 1, 0, 1, 1, 1, 1, 1, 4, 32, 32, xr1, 0);
  float* xp1 = ws.alloc((size_t)32 * 4 * 64 * 64);
  pixel_shuffle2_k<<<cdiv((long)32 * 4 * 64 * 64, 256), 256, 0, stream>>>(xr1, xp1, 32, 4, 32, 32);
  float* xc2 = ws.alloc((size_t)64 * 4 * 64 * 64);
  conv3d_im2col(xp1, 16384, 4096, 64, 1, 32, 4, 64, 64, cbu2_w, cbu2_b, 64,
                1, 3, 3, 1, 1, 1, 0, 1, 1, 1, 1, 1, 4, 64, 64, xc2, 0);
  float* xp2 = ws.alloc((size_t)16 * 4 * 128 * 128);
  pixel_shuffle2_k<<<cdiv((long)16 * 4 * 128 * 128, 256), 256, 0, stream>>>(xc2, xp2, 16, 4, 64, 64);
  float* xlast = ws.alloc((size_t)3 * 4 * 128 * 128);
  conv3d_im2col(xp2, 65536, 16384, 128, 1, 16, 4, 128, 128, last_w, last_b, 3,
                1, 3, 3, 1, 1, 1, 0, 1, 1, 1, 1, 1, 4, 128, 128, xlast, 0);

  // (3,4,H,W) -> (4,3,H,W) into d_out
  cthw_to_tchw_k<<<cdiv((long)3 * 4 * 16384, 256), 256, 0, stream>>>(xlast, (float*)d_out,
                                                                     3, 4, 16384);
}
